// Model_70643622084745
// MI455X (gfx1250) — matrix operand order repaired and checked by reading, not compiled, not run on hardware
//
#include <hip/hip_runtime.h>
#include <stdint.h>

// ---------------------------------------------------------------------------
// CDNA5 (gfx1250) wave32 WMMA implementation of the dual-reprogramming +
// adaptive-gate model. Heavy math through v_wmma_f32_16x16x32_bf16; bf16
// bulk staging through the Tensor Data Mover (tensor_load_to_lds).
// ---------------------------------------------------------------------------

typedef __attribute__((ext_vector_type(16))) __bf16 v16bf;
typedef __attribute__((ext_vector_type(8)))  float  v8f;
typedef __attribute__((ext_vector_type(4)))  unsigned v4u;
typedef __attribute__((ext_vector_type(4)))  int      v4i;
typedef __attribute__((ext_vector_type(8)))  int      v8i;

struct __align__(16) U128 { unsigned x, y, z, w; };
struct __align__(16) F128 { float x, y, z, w; };

union BF16Frag {
    v16bf v;
    U128  q[2];
    unsigned short h[16];
};

__device__ __forceinline__ unsigned short f2bf(float f) {
    unsigned u = __builtin_bit_cast(unsigned, f);
    u += 0x7FFFu + ((u >> 16) & 1u);          // round-to-nearest-even
    return (unsigned short)(u >> 16);
}
__device__ __forceinline__ float bf2f(unsigned short h) {
    return __builtin_bit_cast(float, (unsigned)h << 16);
}

__device__ __forceinline__ v8f wmma_bf16(const BF16Frag& a, const BF16Frag& b, v8f c) {
    return __builtin_amdgcn_wmma_f32_16x16x32_bf16(
        /*neg_a=*/false, a.v, /*neg_b=*/false, b.v,
        /*c_mod=*/(short)0, c, /*reuse_a=*/false, /*reuse_b=*/false);
}

// ---------------------------------------------------------------------------
// Tensor Data Mover: 2-D bf16 tile load Global->LDS.
//   D# group0: count=1, lds_addr, 57-bit global addr, type=2 ("image")
//   D# group1: data_size=1(2B), pad_enable + interval/amount, tensor dims
//              (rows beyond tensor_h read as zero), tile dims, row stride.
// ---------------------------------------------------------------------------
__device__ __forceinline__ void tdm_load_2d_bf16(
    unsigned lds_addr, const void* gaddr,
    unsigned tensor_w, unsigned tensor_h,
    unsigned tile_w, unsigned tile_h,
    unsigned row_stride_elems,
    unsigned pad_interval_code, unsigned pad_amount_code)
{
    unsigned long long ga = (unsigned long long)gaddr;
    v4u g0;
    g0[0] = 1u;                                            // count=1 (user D#)
    g0[1] = lds_addr;                                      // LDS byte address
    g0[2] = (unsigned)(ga & 0xFFFFFFFFu);                  // global_addr[31:0]
    g0[3] = (unsigned)((ga >> 32) & 0x1FFFFFFu)            // global_addr[56:32]
          | (2u << 30);                                    // type=2
    v8i g1;
    g1[0] = (int)((1u << 16)                               // data_size: 2 bytes
                | (1u << 20)                               // pad_enable
                | (pad_interval_code << 22)
                | (pad_amount_code << 25));
    g1[1] = (int)((tensor_w & 0xFFFFu) << 16);                               // dim0[15:0]
    g1[2] = (int)(((tensor_w >> 16) & 0xFFFFu) | ((tensor_h & 0xFFFFu) << 16)); // dim0[31:16] | dim1[15:0]
    g1[3] = (int)(((tensor_h >> 16) & 0xFFFFu) | ((tile_w & 0xFFFFu) << 16));   // dim1[31:16] | tile_dim0
    g1[4] = (int)(tile_h & 0xFFFFu);                       // tile_dim1 (tile_dim2=0)
    g1[5] = (int)row_stride_elems;                         // dim0_stride[31:0]
    g1[6] = 0;                                             // dim0_stride[47:32] | dim1_stride lo
    g1[7] = 0;
    v4i z4 = {0, 0, 0, 0};
#if __has_include(<hip/amd_detail/amd_gfx1250_TDM.h>)
    v8i z8 = {0, 0, 0, 0, 0, 0, 0, 0};
    __builtin_amdgcn_tensor_load_to_lds(g0, g1, z4, z4, z8, 0);
#else
    __builtin_amdgcn_tensor_load_to_lds(g0, g1, z4, z4, 0);
#endif
}

// ---------------------------------------------------------------------------
// Generic bf16 WMMA GEMM:  C(M,N) = act( A(M,K) @ B(K,N) + bias )
//   A: fp32 (converted on the fly, optional concat split A0|A1),
//      or bf16 (staged whole-tile through the TDM).
//   B: fp32 weights (K x N row-major), converted into LDS transposed.
//   Block tile 128x128x32, 256 threads (8 waves), wave tile 32x64.
// ---------------------------------------------------------------------------
template<bool A_BF16, bool RELU, bool OUT_BF16>
__global__ __launch_bounds__(256)
void gemm_wmma(const void* __restrict__ Aptr0, const void* __restrict__ Aptr1,
               int splitK,
               const float* __restrict__ Bw, const float* __restrict__ bias,
               void* __restrict__ Cptr, int M, int N, int K, int lda, int ldb)
{
    __shared__ unsigned short As [128][40];   // [m][k]  row stride 80B (16B mult)
    __shared__ unsigned short BsT[128][40];   // [n][k]  transposed B tile

    const int tid   = threadIdx.x;
    const int wave  = tid >> 5;
    const int lane  = tid & 31;
    const int half  = lane >> 4;
    const int l16   = lane & 15;
    const int waveM = wave >> 1;              // 0..3  (rows of 32)
    const int waveN = wave & 1;               // 0..1  (cols of 64)
    const int tileM = blockIdx.y * 128;
    const int tileN = blockIdx.x * 128;

    v8f acc[2][4];
#pragma unroll
    for (int i = 0; i < 2; ++i)
#pragma unroll
        for (int j = 0; j < 4; ++j)
            acc[i][j] = (v8f){0.f,0.f,0.f,0.f,0.f,0.f,0.f,0.f};

    const int arow = tid >> 1;                // 0..127
    const int acol = (tid & 1) * 16;          // 0 or 16
    const int brow = tid >> 3;                // 0..31
    const int bcol = (tid & 7) * 16;          // 0..112

    const unsigned asLds = (unsigned)(uintptr_t)&As[0][0];

    for (int kb = 0; kb < K; kb += 32) {
        // prefetch next B tile rows (global_prefetch_b8)
        if (kb + 32 < K)
            __builtin_prefetch(Bw + (size_t)(kb + 32 + brow) * ldb + tileN + bcol, 0, 1);

        // ---- stage A tile (128 x 32) ----
        if constexpr (A_BF16) {
            // Tensor Data Mover: 32-wide x 128-row bf16 tile, rows past M zero.
            // LDS pad: 16 DWORDs data + 4 DWORDs pad -> row stride 40 ushorts.
            if (wave == 0)
                tdm_load_2d_bf16(asLds,
                                 (const unsigned short*)Aptr0 + (size_t)tileM * lda + kb,
                                 /*tensor_w=*/32, /*tensor_h=*/(unsigned)(M - tileM),
                                 /*tile_w=*/32, /*tile_h=*/128,
                                 (unsigned)lda,
                                 /*interval 16DW=*/3, /*amount 4DW=*/3);
        } else {
            const int rg = tileM + arow;
            F128 fb[4];
#pragma unroll
            for (int i = 0; i < 4; ++i) fb[i] = (F128){0.f,0.f,0.f,0.f};
            if (rg < M) {
                const int cg = kb + acol;     // splitK is a multiple of 16
                const float* ap = (cg < splitK)
                    ? (const float*)Aptr0 + (size_t)rg * lda + cg
                    : (const float*)Aptr1 + (size_t)rg * lda + (cg - splitK);
#pragma unroll
                for (int i = 0; i < 4; ++i) fb[i] = *(const F128*)(ap + i * 4);
            }
            const float* f = (const float*)fb;
            unsigned* dst = (unsigned*)&As[arow][acol];
#pragma unroll
            for (int i = 0; i < 8; ++i)
                dst[i] = (unsigned)f2bf(f[2*i]) | ((unsigned)f2bf(f[2*i+1]) << 16);
        }
        // ---- stage B tile (32 x 128), transposed into LDS (f32 -> bf16) ----
        {
            const float* bp = Bw + (size_t)(kb + brow) * ldb + tileN + bcol;
            F128 fb[4];
#pragma unroll
            for (int i = 0; i < 4; ++i) fb[i] = *(const F128*)(bp + i * 4);
            const float* f = (const float*)fb;
#pragma unroll
            for (int i = 0; i < 16; ++i)
                BsT[bcol + i][brow] = f2bf(f[i]);
        }
        if constexpr (A_BF16) {
            if (wave == 0) __builtin_amdgcn_s_wait_tensorcnt(0);
        }
        __syncthreads();

        // ---- fragments + 8 WMMAs per wave ----
        BF16Frag a[2], b[4];
#pragma unroll
        for (int i = 0; i < 2; ++i) {
            const unsigned short* ar = &As[waveM * 32 + i * 16 + l16][0];
            a[i].q[0] = *(const U128*)(ar + half * 8);
            a[i].q[1] = *(const U128*)(ar + 16 + half * 8);
        }
#pragma unroll
        for (int j = 0; j < 4; ++j) {
            const unsigned short* br = &BsT[waveN * 64 + j * 16 + l16][0];
            b[j].q[0] = *(const U128*)(br + half * 8);
            b[j].q[1] = *(const U128*)(br + 16 + half * 8);
        }
#pragma unroll
        for (int i = 0; i < 2; ++i)
#pragma unroll
            for (int j = 0; j < 4; ++j)
                acc[i][j] = wmma_bf16(a[i], b[j], acc[i][j]);
        __syncthreads();
    }

    // ---- epilogue: bias (+ReLU) + store ----
#pragma unroll
    for (int i = 0; i < 2; ++i) {
#pragma unroll
        for (int j = 0; j < 4; ++j) {
            const int col = tileN + waveN * 64 + j * 16 + l16;
            const float bs = bias[col];
#pragma unroll
            for (int r = 0; r < 8; ++r) {
                const int row = tileM + waveM * 32 + i * 16 + r + 8 * half;
                if (row < M) {
                    float v = acc[i][j][r] + bs;
                    if constexpr (RELU) v = fmaxf(v, 0.f);
                    if constexpr (OUT_BF16)
                        ((unsigned short*)Cptr)[(size_t)row * N + col] = f2bf(v);
                    else
                        ((float*)Cptr)[(size_t)row * N + col] = v;
                }
            }
        }
    }
}

// ---------------------------------------------------------------------------
// Flash attention over prototypes: per (b,h) workgroup of 4 waves, each wave
// owns 16 query rows, streams S=1000 keys/values in chunks of 32.
// K chunk staged by the TDM (zero-fill past S=1000 + LDS padding); V chunk
// transposed manually (TDM cannot transpose).
// Q:(4096,1024) bf16, K/V:(1000,1024) bf16, O:(4096,1024) bf16.
// ---------------------------------------------------------------------------
__global__ __launch_bounds__(128)
void attn_kernel(const unsigned short* __restrict__ Q,
                 const unsigned short* __restrict__ Kb,
                 const unsigned short* __restrict__ Vb,
                 unsigned short* __restrict__ O)
{
    const int h    = blockIdx.y;
    const int bidx = blockIdx.z;
    const int tid  = threadIdx.x;
    const int wave = tid >> 5;
    const int lane = tid & 31;
    const int half = lane >> 4;
    const int l16  = lane & 15;

    __shared__ unsigned short Ks [32][72];        // K chunk [s][e], 144B stride
    __shared__ unsigned short Vs [64][40];        // V chunk transposed [e][s]
    __shared__ unsigned short Pls[4][16][40];     // per-wave P staging (C->A relayout)

    const int rowQ = bidx * 512 + blockIdx.x * 64 + wave * 16;
    const unsigned ksLds = (unsigned)(uintptr_t)&Ks[0][0];

    // preload this wave's Q A-fragments (16x64 -> two 16x32 frags), kept in VGPRs
    BF16Frag aQ[2];
    {
        const unsigned short* qrow = Q + (size_t)(rowQ + l16) * 1024 + h * 64;
#pragma unroll
        for (int c = 0; c < 2; ++c) {
            aQ[c].q[0] = *(const U128*)(qrow + c * 32 + half * 8);
            aQ[c].q[1] = *(const U128*)(qrow + c * 32 + 16 + half * 8);
        }
    }

    v8f accO[4];
#pragma unroll
    for (int t = 0; t < 4; ++t) accO[t] = (v8f){0.f,0.f,0.f,0.f,0.f,0.f,0.f,0.f};
    float rmax[8], rsum[8];
#pragma unroll
    for (int r = 0; r < 8; ++r) { rmax[r] = -1e30f; rsum[r] = 0.f; }

    const float scale = 0.125f;   // 1/sqrt(64)

    for (int s0 = 0; s0 < 1024; s0 += 32) {
        // ---- stage K chunk via TDM: 64-wide x 32-row, rows past S zeroed,
        //      LDS pad 32DW+4DW -> row stride 72 ushorts ----
        if (wave == 0)
            tdm_load_2d_bf16(ksLds,
                             Kb + (size_t)s0 * 1024 + h * 64,
                             /*tensor_w=*/64, /*tensor_h=*/(unsigned)(1000 - s0),
                             /*tile_w=*/64, /*tile_h=*/32,
                             /*stride=*/1024,
                             /*interval 32DW=*/4, /*amount 4DW=*/3);
        // ---- stage V chunk transposed (manual) ----
        {
            const int r  = tid >> 2;            // 0..31 (s within chunk)
            const int cb = (tid & 3) * 16;      // 0,16,32,48 (e)
            const int sg = s0 + r;
            U128 v0 = {0,0,0,0}, v1 = {0,0,0,0};
            if (sg < 1000) {
                const unsigned short* vp = Vb + (size_t)sg * 1024 + h * 64 + cb;
                v0 = *(const U128*)(vp);  v1 = *(const U128*)(vp + 8);
            }
            unsigned short tmp[16];
            *(U128*)&tmp[0] = v0; *(U128*)&tmp[8] = v1;
#pragma unroll
            for (int i = 0; i < 16; ++i) Vs[cb + i][r] = tmp[i];
        }
        if (wave == 0) __builtin_amdgcn_s_wait_tensorcnt(0);
        __syncthreads();

        // ---- scores: S_tile(16x32) = Q(16x64) @ K_chunk^T (batched frags) ----
        BF16Frag bK[4];
#pragma unroll
        for (int c = 0; c < 2; ++c) {
            const unsigned short* krow = &Ks[c * 16 + l16][0];
            bK[c*2+0].q[0] = *(const U128*)(krow + half * 8);
            bK[c*2+0].q[1] = *(const U128*)(krow + 16 + half * 8);
            bK[c*2+1].q[0] = *(const U128*)(krow + 32 + half * 8);
            bK[c*2+1].q[1] = *(const U128*)(krow + 32 + 16 + half * 8);
        }
        v8f sc[2];
#pragma unroll
        for (int c = 0; c < 2; ++c) {
            v8f z = (v8f){0.f,0.f,0.f,0.f,0.f,0.f,0.f,0.f};
            z = wmma_bf16(aQ[0], bK[c*2+0], z);
            z = wmma_bf16(aQ[1], bK[c*2+1], z);
            sc[c] = z;
        }
        // scale + mask out-of-range prototypes (S=1000, padded to 1024)
#pragma unroll
        for (int c = 0; c < 2; ++c) {
            const bool valid = (s0 + c * 16 + l16) < 1000;
#pragma unroll
            for (int r = 0; r < 8; ++r)
                sc[c][r] = valid ? sc[c][r] * scale : -1e30f;
        }
        // ---- online softmax across 16-lane groups ----
        float corr[8];
#pragma unroll
        for (int r = 0; r < 8; ++r) {
            float m = fmaxf(sc[0][r], sc[1][r]);
#pragma unroll
            for (int msk = 8; msk >= 1; msk >>= 1)
                m = fmaxf(m, __shfl_xor(m, msk, 32));
            const float nm = fmaxf(rmax[r], m);
            corr[r] = __expf(rmax[r] - nm);
            rmax[r] = nm;
        }
#pragma unroll
        for (int r = 0; r < 8; ++r) {
            const float p0 = __expf(sc[0][r] - rmax[r]);
            const float p1 = __expf(sc[1][r] - rmax[r]);
            sc[0][r] = p0; sc[1][r] = p1;
            float rs = p0 + p1;
#pragma unroll
            for (int msk = 8; msk >= 1; msk >>= 1)
                rs += __shfl_xor(rs, msk, 32);
            rsum[r] = rsum[r] * corr[r] + rs;
#pragma unroll
            for (int t = 0; t < 4; ++t) accO[t][r] *= corr[r];
        }
        // ---- relayout P (C-layout -> A-layout) through per-wave LDS ----
#pragma unroll
        for (int c = 0; c < 2; ++c)
#pragma unroll
            for (int r = 0; r < 8; ++r)
                Pls[wave][r + 8 * half][c * 16 + l16] = f2bf(sc[c][r]);
        BF16Frag aP;
        {
            const unsigned short* prow = &Pls[wave][l16][0];
            aP.q[0] = *(const U128*)(prow + half * 8);
            aP.q[1] = *(const U128*)(prow + 16 + half * 8);
        }
        // ---- accO += P(16x32) @ V_chunk(32x64) ----
        BF16Frag bV[4];
#pragma unroll
        for (int t = 0; t < 4; ++t) {
            const unsigned short* vrow = &Vs[t * 16 + l16][0];
            bV[t].q[0] = *(const U128*)(vrow + half * 8);
            bV[t].q[1] = *(const U128*)(vrow + 16 + half * 8);
        }
#pragma unroll
        for (int t = 0; t < 4; ++t)
            accO[t] = wmma_bf16(aP, bV[t], accO[t]);
        __syncthreads();
    }

    // ---- normalize + store bf16 ----
#pragma unroll
    for (int t = 0; t < 4; ++t)
#pragma unroll
        for (int r = 0; r < 8; ++r) {
            const float o  = accO[t][r] / rsum[r];
            const int row  = rowQ + r + 8 * half;
            O[(size_t)row * 1024 + h * 64 + t * 16 + l16] = f2bf(o);
        }
}

// ---------------------------------------------------------------------------
// gate[row] = sigmoid( h[row,:] . W2 + b2 )   (wave per row)
// ---------------------------------------------------------------------------
__global__ __launch_bounds__(256)
void gate_kernel(const unsigned short* __restrict__ H, const float* __restrict__ W2,
                 const float* __restrict__ b2, float* __restrict__ gate, int rows)
{
    const int row  = blockIdx.x * 8 + (threadIdx.x >> 5);
    const int lane = threadIdx.x & 31;
    if (row >= rows) return;
    const unsigned short* hp = H + (size_t)row * 1024;
    float s = 0.f;
    for (int i = lane; i < 1024; i += 32)
        s += bf2f(hp[i]) * W2[i];
#pragma unroll
    for (int m = 16; m >= 1; m >>= 1) s += __shfl_xor(s, m, 32);
    if (lane == 0) gate[row] = 1.f / (1.f + __expf(-(s + b2[0])));
}

// ---------------------------------------------------------------------------
// out = gate * sem_t(in d_out) + (1-gate) * sem_d     (float4 per thread)
// ---------------------------------------------------------------------------
__global__ __launch_bounds__(256)
void blend_kernel(float* __restrict__ out, const float* __restrict__ semd,
                  const float* __restrict__ gate)
{
    const size_t i = (size_t)blockIdx.x * blockDim.x + threadIdx.x;  // float4 idx
    const int row = (int)(i >> 10);                                   // 4096/4 per row
    const float g = gate[row];
    F128 a = ((const F128*)out)[i];
    F128 d = ((const F128*)semd)[i];
    F128 r;
    r.x = g * a.x + (1.f - g) * d.x;
    r.y = g * a.y + (1.f - g) * d.y;
    r.z = g * a.z + (1.f - g) * d.z;
    r.w = g * a.w + (1.f - g) * d.w;
    ((F128*)out)[i] = r;
}

// ---------------------------------------------------------------------------
extern "C" void kernel_launch(void* const* d_in, const int* in_sizes, int n_in,
                              void* d_out, int out_size, void* d_ws, size_t ws_size,
                              hipStream_t stream)
{
    const float* Xt   = (const float*)d_in[0];   // (4096,1024)
    const float* Xd   = (const float*)d_in[1];
    const float* Pt   = (const float*)d_in[2];   // (1000,4096)
    const float* Pd   = (const float*)d_in[3];
    const float* tWq  = (const float*)d_in[4];   const float* tbq = (const float*)d_in[5];
    const float* tWk  = (const float*)d_in[6];   const float* tbk = (const float*)d_in[7];
    const float* tWv  = (const float*)d_in[8];   const float* tbv = (const float*)d_in[9];
    const float* tWo  = (const float*)d_in[10];  const float* tbo = (const float*)d_in[11];
    const float* dWq  = (const float*)d_in[12];  const float* dbq = (const float*)d_in[13];
    const float* dWk  = (const float*)d_in[14];  const float* dbk = (const float*)d_in[15];
    const float* dWv  = (const float*)d_in[16];  const float* dbv = (const float*)d_in[17];
    const float* dWo  = (const float*)d_in[18];  const float* dbo = (const float*)d_in[19];
    const float* gW1  = (const float*)d_in[20];  const float* gb1 = (const float*)d_in[21];
    const float* gW2  = (const float*)d_in[22];  const float* gb2 = (const float*)d_in[23];

    char* ws = (char*)d_ws;
    size_t off = 0;
    auto wsalloc = [&](size_t bytes) -> void* {
        void* p = ws + off;
        off += (bytes + 255) & ~(size_t)255;
        return p;
    };
    unsigned short* Qt  = (unsigned short*)wsalloc((size_t)4096 * 1024 * 2);
    unsigned short* Qd  = (unsigned short*)wsalloc((size_t)4096 * 1024 * 2);
    unsigned short* Kt  = (unsigned short*)wsalloc((size_t)1000 * 1024 * 2);
    unsigned short* Vt  = (unsigned short*)wsalloc((size_t)1000 * 1024 * 2);
    unsigned short* Kd  = (unsigned short*)wsalloc((size_t)1000 * 1024 * 2);
    unsigned short* Vd  = (unsigned short*)wsalloc((size_t)1000 * 1024 * 2);
    unsigned short* At  = (unsigned short*)wsalloc((size_t)4096 * 1024 * 2);
    unsigned short* Ad  = (unsigned short*)wsalloc((size_t)4096 * 1024 * 2);
    unsigned short* Hh  = (unsigned short*)wsalloc((size_t)4096 * 1024 * 2);
    float*          gat = (float*)wsalloc((size_t)4096 * 4);
    float*          smd = (float*)wsalloc((size_t)4096 * 4096 * 4);

    const dim3 blk(256);

    // Q projections: (4096x1024) = X @ Wq + bq
    gemm_wmma<false, false, true><<<dim3(8, 32), blk, 0, stream>>>(
        Xt, Xt, 1024, tWq, tbq, Qt, 4096, 1024, 1024, 1024, 1024);
    gemm_wmma<false, false, true><<<dim3(8, 32), blk, 0, stream>>>(
        Xd, Xd, 1024, dWq, dbq, Qd, 4096, 1024, 1024, 1024, 1024);

    // K/V projections: (1000x1024) = P @ W + b
    gemm_wmma<false, false, true><<<dim3(8, 8), blk, 0, stream>>>(
        Pt, Pt, 4096, tWk, tbk, Kt, 1000, 1024, 4096, 4096, 1024);
    gemm_wmma<false, false, true><<<dim3(8, 8), blk, 0, stream>>>(
        Pt, Pt, 4096, tWv, tbv, Vt, 1000, 1024, 4096, 4096, 1024);
    gemm_wmma<false, false, true><<<dim3(8, 8), blk, 0, stream>>>(
        Pd, Pd, 4096, dWk, dbk, Kd, 1000, 1024, 4096, 4096, 1024);
    gemm_wmma<false, false, true><<<dim3(8, 8), blk, 0, stream>>>(
        Pd, Pd, 4096, dWv, dbv, Vd, 1000, 1024, 4096, 4096, 1024);

    // gate hidden: (4096x1024) = relu([Xt|Xd] @ gW1 + gb1)   (concat, K=2048)
    gemm_wmma<false, true, true><<<dim3(8, 32), blk, 0, stream>>>(
        Xt, Xd, 1024, gW1, gb1, Hh, 4096, 1024, 2048, 1024, 1024);

    // attention (flash over 1000 prototypes), per (qtile, head, batch)
    attn_kernel<<<dim3(8, 16, 8), dim3(128), 0, stream>>>(Qt, Kt, Vt, At);
    attn_kernel<<<dim3(8, 16, 8), dim3(128), 0, stream>>>(Qd, Kd, Vd, Ad);

    // output projections: sem_t -> d_out, sem_d -> ws (A is bf16 -> TDM staging)
    gemm_wmma<true, false, false><<<dim3(32, 32), blk, 0, stream>>>(
        At, At, 1024, tWo, tbo, d_out, 4096, 4096, 1024, 1024, 4096);
    gemm_wmma<true, false, false><<<dim3(32, 32), blk, 0, stream>>>(
        Ad, Ad, 1024, dWo, dbo, smd, 4096, 4096, 1024, 1024, 4096);

    // gate + blend
    gate_kernel<<<dim3(512), blk, 0, stream>>>(Hh, gW2, gb2, gat, 4096);
    blend_kernel<<<dim3(16384), blk, 0, stream>>>((float*)d_out, smd, gat);
}
